// MoE_31121333027446
// MI455X (gfx1250) — compile-verified, matrix operand order checked by reading, not run on hardware
//
#include <hip/hip_runtime.h>

// ---------------- constants ----------------
#define D_MODEL 1024
#define D_FF    4096
#define NEXP    8
#define NTOK    8192      // BATCH * SEQ
#define MT      32        // token rows per block (2 WMMA row-tiles)
#define CF      256       // ff chunk size held in LDS

typedef __bf16 v8bf  __attribute__((ext_vector_type(8)));
typedef __bf16 v16bf __attribute__((ext_vector_type(16)));
typedef float  v8f   __attribute__((ext_vector_type(8)));

static __device__ __forceinline__ v16bf cat16(v8bf lo, v8bf hi) {
  return __builtin_shufflevector(lo, hi, 0,1,2,3,4,5,6,7,8,9,10,11,12,13,14,15);
}

// ---------------- transpose + fp32->bf16 convert ----------------
// in:  [E][R][C] fp32, out: [E][C][R] bf16.  block (32,8), grid (C/32, R/32, E)
__global__ __launch_bounds__(256)
void transpose_cvt_kernel(const float* __restrict__ in, __bf16* __restrict__ out,
                          int R, int C) {
  __shared__ float tile[32][33];
  const int e  = blockIdx.z;
  const int c0 = blockIdx.x * 32;
  const int r0 = blockIdx.y * 32;
  const float* inp  = in  + (size_t)e * R * C;
  __bf16*      outp = out + (size_t)e * R * C;
  const int tx = threadIdx.x, ty = threadIdx.y;
#pragma unroll
  for (int i = 0; i < 32; i += 8)
    tile[ty + i][tx] = inp[(size_t)(r0 + ty + i) * C + (c0 + tx)];
  __syncthreads();
#pragma unroll
  for (int i = 0; i < 32; i += 8)
    outp[(size_t)(c0 + ty + i) * R + (r0 + tx)] = (__bf16)tile[tx][ty + i];
}

// ---------------- router: logits, top-2, gather lists, x->bf16 ----------------
// one block (256 threads) per token
__global__ __launch_bounds__(256)
void router_gather_kernel(const float* __restrict__ x,
                          const float* __restrict__ rw,   // [D_MODEL][NEXP]
                          const float* __restrict__ rb,   // [NEXP]
                          __bf16* __restrict__ xb,        // [NTOK][D_MODEL]
                          int*    __restrict__ cnt,       // [NEXP]
                          int*    __restrict__ toks,      // [NEXP][NTOK]
                          float*  __restrict__ gat) {     // [NEXP][NTOK]
  const int token = blockIdx.x;
  const int t     = threadIdx.x;
  const float4 xv = ((const float4*)(x + (size_t)token * D_MODEL))[t];

  // write bf16 copy of x row
  __bf16* xo = xb + (size_t)token * D_MODEL + t * 4;
  xo[0] = (__bf16)xv.x; xo[1] = (__bf16)xv.y;
  xo[2] = (__bf16)xv.z; xo[3] = (__bf16)xv.w;

  const float* r0 = rw + (size_t)(t * 4) * NEXP;
  float acc[NEXP];
#pragma unroll
  for (int e = 0; e < NEXP; ++e)
    acc[e] = xv.x * r0[e] + xv.y * r0[NEXP + e] + xv.z * r0[2 * NEXP + e]
           + xv.w * r0[3 * NEXP + e];

  __shared__ float sred[NEXP * 256];
#pragma unroll
  for (int e = 0; e < NEXP; ++e) sred[e * 256 + t] = acc[e];
  __syncthreads();
  for (int s = 128; s > 0; s >>= 1) {
    if (t < s) {
#pragma unroll
      for (int e = 0; e < NEXP; ++e) sred[e * 256 + t] += sred[e * 256 + t + s];
    }
    __syncthreads();
  }
  if (t == 0) {
    float lg[NEXP];
#pragma unroll
    for (int e = 0; e < NEXP; ++e) lg[e] = sred[e * 256] + rb[e];
    int e0 = 0;
#pragma unroll
    for (int e = 1; e < NEXP; ++e) if (lg[e] > lg[e0]) e0 = e;
    int e1 = (e0 == 0) ? 1 : 0;
#pragma unroll
    for (int e = 0; e < NEXP; ++e)
      if (e != e0 && lg[e] > lg[e1]) e1 = e;
    const float g0 = 1.f / (1.f + __expf(lg[e1] - lg[e0]));
    const float g1 = 1.f - g0;
    int p0 = atomicAdd(&cnt[e0], 1);
    toks[e0 * NTOK + p0] = token; gat[e0 * NTOK + p0] = g0;
    int p1 = atomicAdd(&cnt[e1], 1);
    toks[e1 * NTOK + p1] = token; gat[e1 * NTOK + p1] = g1;
  }
}

// ---------------- fused expert FFN over gathered 32-token tiles ----------------
// grid (NEXP, NTOK/MT), block 256 (8 waves)
__global__ __launch_bounds__(256, 1)
void moe_expert_kernel(const __bf16* __restrict__ xb,   // [NTOK][D_MODEL]
                       const __bf16* __restrict__ W1t,  // [E][D_FF][D_MODEL]
                       const __bf16* __restrict__ W2t,  // [E][D_MODEL][D_FF]
                       const float*  __restrict__ b1,   // [E][D_FF]
                       const float*  __restrict__ b2,   // [E][D_MODEL]
                       const int*    __restrict__ cnt,
                       const int*    __restrict__ toks,
                       const float*  __restrict__ gat,
                       float*        __restrict__ out) {
  const int e    = blockIdx.x;
  const int n    = cnt[e];
  const int row0 = blockIdx.y * MT;
  if (row0 >= n) return;                       // block-uniform: EXEC stays full

  __shared__ __bf16 xg[MT][D_MODEL];           // 64 KB gathered activations
  __shared__ __bf16 hs[MT][CF];                // 16 KB hidden chunk
  __shared__ int    ttok[MT];
  __shared__ float  tgate[MT];

  const int tid  = threadIdx.x;
  const int wave = tid >> 5;
  const int lane = tid & 31;
  const int half = lane >> 4;                  // which 16-lane half
  const int lrow = lane & 15;

  if (tid < MT) {
    const int r = row0 + tid;
    if (r < n) { ttok[tid] = toks[e * NTOK + r]; tgate[tid] = gat[e * NTOK + r]; }
    else       { ttok[tid] = -1;                tgate[tid] = 0.f; }
  }
  __syncthreads();

  // gather MT token rows of bf16 x into LDS (v8 chunks)
  for (int i = tid; i < MT * (D_MODEL / 8); i += 256) {
    const int r  = i / (D_MODEL / 8);
    const int c8 = i % (D_MODEL / 8);
    const int tok = ttok[r];
    v8bf v;
    if (tok >= 0) v = *(const v8bf*)(xb + (size_t)tok * D_MODEL + c8 * 8);
    else {
#pragma unroll
      for (int j = 0; j < 8; ++j) v[j] = (__bf16)0.f;
    }
    *(v8bf*)(&xg[r][c8 * 8]) = v;
  }
  __syncthreads();

  const __bf16* W1e = W1t + (size_t)e * D_FF * D_MODEL;
  const __bf16* W2e = W2t + (size_t)e * D_MODEL * D_FF;

  v8f yacc[2][8];                              // [row-tile][col-tile], 128 VGPR
#pragma unroll
  for (int rt = 0; rt < 2; ++rt)
#pragma unroll
    for (int t = 0; t < 8; ++t)
#pragma unroll
      for (int j = 0; j < 8; ++j) yacc[rt][t][j] = 0.f;

  for (int ffc = 0; ffc < D_FF / CF; ++ffc) {
    const int fbase = ffc * CF;

    // ---- GEMM1: h[MT, CF] = relu(xg @ W1[:, fbase:fbase+CF] + b1) ----
    // wave covers CF/8 = 32 columns -> 2 col-tiles x 2 row-tiles
    v8f hacc[2][2];
#pragma unroll
    for (int rt = 0; rt < 2; ++rt)
#pragma unroll
      for (int ct = 0; ct < 2; ++ct)
#pragma unroll
        for (int j = 0; j < 8; ++j) hacc[rt][ct][j] = 0.f;

    for (int kk = 0; kk < D_MODEL; kk += 32) {
      v16bf A[2];
#pragma unroll
      for (int rt = 0; rt < 2; ++rt) {
        const v8bf alo = *(const v8bf*)(&xg[rt * 16 + lrow][kk + half * 8]);
        const v8bf ahi = *(const v8bf*)(&xg[rt * 16 + lrow][kk + 16 + half * 8]);
        A[rt] = cat16(alo, ahi);
      }
#pragma unroll
      for (int ct = 0; ct < 2; ++ct) {
        const int f = fbase + wave * 32 + ct * 16 + lrow;
        const __bf16* bp = W1e + (size_t)f * D_MODEL + kk + half * 8;
        const v16bf B = cat16(*(const v8bf*)bp, *(const v8bf*)(bp + 16));
#pragma unroll
        for (int rt = 0; rt < 2; ++rt)
          hacc[rt][ct] = __builtin_amdgcn_wmma_f32_16x16x32_bf16(
              false, A[rt], false, B, (short)0, hacc[rt][ct], false, false);
      }
    }
#pragma unroll
    for (int ct = 0; ct < 2; ++ct) {
      const int fcol = wave * 32 + ct * 16 + lrow;
      const float bias = b1[e * D_FF + fbase + fcol];
#pragma unroll
      for (int rt = 0; rt < 2; ++rt)
#pragma unroll
        for (int j = 0; j < 8; ++j) {
          float hv = fmaxf(hacc[rt][ct][j] + bias, 0.f);
          hs[rt * 16 + j + half * 8][fcol] = (__bf16)hv;
        }
    }
    __syncthreads();

    // ---- GEMM2: y[MT, D_MODEL] += h[MT, CF] @ W2[fbase:fbase+CF, :] ----
    for (int kk = 0; kk < CF; kk += 32) {
      v16bf A[2];
#pragma unroll
      for (int rt = 0; rt < 2; ++rt) {
        const v8bf alo = *(const v8bf*)(&hs[rt * 16 + lrow][kk + half * 8]);
        const v8bf ahi = *(const v8bf*)(&hs[rt * 16 + lrow][kk + 16 + half * 8]);
        A[rt] = cat16(alo, ahi);
      }
#pragma unroll
      for (int t = 0; t < 8; ++t) {
        const int d = wave * 128 + t * 16 + lrow;
        const __bf16* bp = W2e + (size_t)d * D_FF + fbase + kk + half * 8;
        const v16bf B = cat16(*(const v8bf*)bp, *(const v8bf*)(bp + 16));
#pragma unroll
        for (int rt = 0; rt < 2; ++rt)
          yacc[rt][t] = __builtin_amdgcn_wmma_f32_16x16x32_bf16(
              false, A[rt], false, B, (short)0, yacc[rt][t], false, false);
      }
    }
    __syncthreads();
  }

  // ---- epilogue: +b2, scale by gate, scatter-accumulate ----
#pragma unroll
  for (int t = 0; t < 8; ++t) {
    const int d = wave * 128 + t * 16 + lrow;
    const float bias = b2[e * D_MODEL + d];
#pragma unroll
    for (int rt = 0; rt < 2; ++rt)
#pragma unroll
      for (int j = 0; j < 8; ++j) {
        const int m   = rt * 16 + j + half * 8;
        const int tok = ttok[m];
        if (tok >= 0) {
          const float val = (yacc[rt][t][j] + bias) * tgate[m];
          atomicAdd(out + (size_t)tok * D_MODEL + d, val);
        }
      }
  }
}

// ---------------- launch ----------------
extern "C" void kernel_launch(void* const* d_in, const int* in_sizes, int n_in,
                              void* d_out, int out_size, void* d_ws, size_t ws_size,
                              hipStream_t stream) {
  (void)in_sizes; (void)n_in; (void)ws_size;
  const float* x   = (const float*)d_in[0];
  const float* rw  = (const float*)d_in[1];
  const float* rb  = (const float*)d_in[2];
  const float* W1  = (const float*)d_in[3];
  const float* b1  = (const float*)d_in[4];
  const float* W2  = (const float*)d_in[5];
  const float* b2  = (const float*)d_in[6];
  float* out = (float*)d_out;

  char* ws = (char*)d_ws;
  const size_t szW  = (size_t)NEXP * D_FF * D_MODEL * sizeof(__bf16); // 64 MB
  __bf16* W1t = (__bf16*)(ws);
  __bf16* W2t = (__bf16*)(ws + szW);
  __bf16* xb  = (__bf16*)(ws + 2 * szW);
  const size_t szXB = (size_t)NTOK * D_MODEL * sizeof(__bf16);        // 16 MB
  int*   cnt  = (int*)  (ws + 2 * szW + szXB);
  int*   toks = (int*)  (ws + 2 * szW + szXB + 256);
  float* gat  = (float*)(ws + 2 * szW + szXB + 256 + (size_t)NEXP * NTOK * 4);

  hipMemsetAsync(cnt, 0, NEXP * sizeof(int), stream);
  hipMemsetAsync(out, 0, (size_t)out_size * sizeof(float), stream);

  // W1 [E][D_MODEL][D_FF] -> W1t [E][D_FF][D_MODEL]
  transpose_cvt_kernel<<<dim3(D_FF / 32, D_MODEL / 32, NEXP), dim3(32, 8), 0, stream>>>(
      W1, W1t, D_MODEL, D_FF);
  // W2 [E][D_FF][D_MODEL] -> W2t [E][D_MODEL][D_FF]
  transpose_cvt_kernel<<<dim3(D_MODEL / 32, D_FF / 32, NEXP), dim3(32, 8), 0, stream>>>(
      W2, W2t, D_FF, D_MODEL);

  router_gather_kernel<<<dim3(NTOK), dim3(256), 0, stream>>>(
      x, rw, rb, xb, cnt, toks, gat);

  moe_expert_kernel<<<dim3(NEXP, NTOK / MT), dim3(256), 0, stream>>>(
      xb, W1t, W2t, b1, b2, cnt, toks, gat, out);
}
